// TransformerDynamics_33191507263984
// MI455X (gfx1250) — matrix operand order repaired and checked by reading, not compiled, not run on hardware
//
#include <hip/hip_runtime.h>
#include <math.h>

typedef __attribute__((ext_vector_type(16))) __bf16 v16bf;
typedef __attribute__((ext_vector_type(8)))  float  v8f;
typedef __attribute__((ext_vector_type(4)))  uint   uint32x4;
typedef __attribute__((ext_vector_type(8)))  int    int32x8;
typedef __attribute__((ext_vector_type(4)))  int    int32x4;

#define B_   16
#define T_   512
#define E_   1024
#define NH_  16
#define HD_  64
#define L_   4
#define HID_ 4096
#define ROWS 8192   // B_*T_

// ---------------- Tensor Data Mover: 2D bf16 tile -> LDS (D# per ISA §8) ------
// lds_byte: byte offset of tile start in LDS. gptr: global tile start.
// tile_d0 elems contiguous per row, tile_d1 rows, stride0 elems between rows.
// pad_int: code (2<<c DWORDs before pad), pad_amt: code (c+1 DWORDs of pad).
__device__ __forceinline__ void tdm_load_2d(uint lds_byte, const void* gptr,
                                            uint tile_d0, uint tile_d1,
                                            uint tens_d0, uint tens_d1,
                                            uint stride0, uint pad_int, uint pad_amt) {
  uint64_t g = (uint64_t)(uintptr_t)gptr;
  uint32x4 g0 = { 1u,                                   // count=1, user mode
                  lds_byte,                             // lds_addr
                  (uint)g,                              // global_addr[31:0]
                  (uint)((g >> 32) & 0x1ffffffu) | (2u << 30) };  // ga[56:32], type=2
  int32x8 g1 = {
    (int)((1u << 16) | (1u << 20) | (pad_int << 22) | (pad_amt << 25)), // 2B, pad_en
    (int)((tens_d0 & 0xffffu) << 16),                                   // dim0[15:0]
    (int)(((tens_d0 >> 16) & 0xffffu) | ((tens_d1 & 0xffffu) << 16)),   // dim0hi,dim1lo
    (int)(((tens_d1 >> 16) & 0xffffu) | (tile_d0 << 16)),               // dim1hi,tile0
    (int)(tile_d1 & 0xffffu),                                           // tile1, tile2=0
    (int)stride0, 0, 0 };                                               // dim0 stride
  int32x4 gz4 = {0, 0, 0, 0};
  int32x8 gz8 = {0, 0, 0, 0, 0, 0, 0, 0};
  __builtin_amdgcn_tensor_load_to_lds(g0, g1, gz4, gz4, gz8, 0);
}

// ---------------- fragment loaders (LDS -> VGPR, ISA §7.12.2 layouts) ----------
__device__ __forceinline__ v16bf load_frag_a(const __bf16* row, int kh) {
  union { v16bf v; uint4 u[2]; } t;
  t.u[0] = *(const uint4*)(row + kh * 8);
  t.u[1] = *(const uint4*)(row + 16 + kh * 8);
  return t.v;
}
__device__ __forceinline__ v16bf load_frag_b(const __bf16* row, int kh) {
  union { v16bf v; uint4 u[2]; } t;
  t.u[0] = *(const uint4*)(row + kh * 8);
  t.u[1] = *(const uint4*)(row + 16 + kh * 8);
  return t.v;
}

// ---------------- setup kernels -----------------------------------------------
// weights fp32 [K][N] -> bf16 [N][K]
__global__ void wtrans_kernel(const float* __restrict__ in, __bf16* __restrict__ out,
                              int K, int N) {
  int i = blockIdx.x * 256 + threadIdx.x;
  if (i >= K * N) return;
  int k = i / N, n = i % N;
  out[(size_t)n * K + k] = (__bf16)in[i];
}

__global__ void concat_kernel(const float* __restrict__ z, const float* __restrict__ a,
                              const float* __restrict__ t, __bf16* __restrict__ xin) {
  int i = blockIdx.x * 256 + threadIdx.x;           // 8192*672 total, exact grid
  int row = i / 672, c = i % 672;
  float v;
  if (c < 512)      v = z[(size_t)row * 512 + c];
  else if (c < 576) v = a[(size_t)row * 64 + (c - 512)];
  else              v = t[(size_t)row * 96 + (c - 576)];
  xin[i] = (__bf16)v;
}

__global__ void addpos_kernel(float* __restrict__ x, const float* __restrict__ pos) {
  int i = blockIdx.x * 256 + threadIdx.x;           // 8192*1024 total, exact grid
  int row = i >> 10, c = i & 1023;
  int tt = row & (T_ - 1);
  x[i] += pos[(size_t)tt * 1024 + c];
}

// LN: fp32 in -> bf16 out (feeds WMMA GEMMs)
__global__ void ln_kernel(const float* __restrict__ x, const float* __restrict__ w,
                          const float* __restrict__ b, __bf16* __restrict__ y) {
  __shared__ float s1[256], s2[256];
  int row = blockIdx.x, tid = threadIdx.x;
  const float* xr = x + (size_t)row * 1024;
  float sum = 0.f, sq = 0.f;
  for (int i = tid; i < 1024; i += 256) { float v = xr[i]; sum += v; sq += v * v; }
  s1[tid] = sum; s2[tid] = sq; __syncthreads();
  for (int off = 128; off > 0; off >>= 1) {
    if (tid < off) { s1[tid] += s1[tid + off]; s2[tid] += s2[tid + off]; }
    __syncthreads();
  }
  float mu = s1[0] * (1.f / 1024.f);
  float var = s2[0] * (1.f / 1024.f) - mu * mu;
  float rs = rsqrtf(var + 1e-5f);
  for (int i = tid; i < 1024; i += 256)
    y[(size_t)row * 1024 + i] = (__bf16)((xr[i] - mu) * rs * w[i] + b[i]);
}

// ---------------- TDM-fed, double-buffered WMMA GEMM --------------------------
// A: MxK bf16 row-major. Bt: NxK bf16 (pre-transposed). M%128==0,N%128==0,K%32==0.
// D = act(A@B + bias) (+Res) as fp32, or bf16 to Dbf when OUTBF.
template <int BIAS, int RES, int GELU_, int OUTBF>
__global__ __launch_bounds__(256)
void gemm_kernel(const __bf16* __restrict__ A, const __bf16* __restrict__ Bt,
                 const float* __restrict__ bias, const float* __restrict__ Res,
                 float* __restrict__ D, __bf16* __restrict__ Dbf,
                 int M, int N, int K) {
  __shared__ alignas(16) __bf16 As[2][128][40];   // TDM pads 64B rows to 80B
  __shared__ alignas(16) __bf16 Bs[2][128][40];
  const int tid = threadIdx.x;
  const int lane = tid & 31, w = tid >> 5;
  const int wm = w >> 2, wn = w & 3;              // 2x4 wave grid
  const int lr = lane & 15, kh = lane >> 4;
  const int m0 = blockIdx.y * 128, n0 = blockIdx.x * 128;

  v8f zero = {0.f, 0.f, 0.f, 0.f, 0.f, 0.f, 0.f, 0.f};
  v8f acc[4][2];
#pragma unroll
  for (int i = 0; i < 4; ++i)
#pragma unroll
    for (int j = 0; j < 2; ++j) acc[i][j] = zero;

  const int nk = K >> 5;
  const uint ldsA0 = (uint)(uintptr_t)&As[0][0][0];
  const uint ldsB0 = (uint)(uintptr_t)&Bs[0][0][0];
  const uint bufBytes = 128u * 40u * 2u;          // 10240

  if (w == 0) {                                   // prologue: stage k-step 0
    tdm_load_2d(ldsA0, A + (size_t)m0 * K, 32, 128, (uint)K, (uint)(M - m0), (uint)K, 3, 3);
    tdm_load_2d(ldsB0, Bt + (size_t)n0 * K, 32, 128, (uint)K, (uint)(N - n0), (uint)K, 3, 3);
  }

  for (int i = 0; i < nk; ++i) {
    const int buf = i & 1;
    if (w == 0) {                                 // prefetch k-step i+1, wait step i
      if (i + 1 < nk) {
        const int k1 = (i + 1) << 5;
        tdm_load_2d(ldsA0 + (uint)(buf ^ 1) * bufBytes, A + (size_t)m0 * K + k1,
                    32, 128, (uint)(K - k1), (uint)(M - m0), (uint)K, 3, 3);
        tdm_load_2d(ldsB0 + (uint)(buf ^ 1) * bufBytes, Bt + (size_t)n0 * K + k1,
                    32, 128, (uint)(K - k1), (uint)(N - n0), (uint)K, 3, 3);
        __builtin_amdgcn_s_wait_tensorcnt(2);     // step i done; i+1 in flight
      } else {
        __builtin_amdgcn_s_wait_tensorcnt(0);
      }
    }
    __syncthreads();

    v16bf af[4], bf[2];
#pragma unroll
    for (int ms = 0; ms < 4; ++ms) af[ms] = load_frag_a(&As[buf][wm * 64 + ms * 16 + lr][0], kh);
#pragma unroll
    for (int ns = 0; ns < 2; ++ns) bf[ns] = load_frag_b(&Bs[buf][wn * 32 + ns * 16 + lr][0], kh);
#pragma unroll
    for (int ms = 0; ms < 4; ++ms)
#pragma unroll
      for (int ns = 0; ns < 2; ++ns)
        acc[ms][ns] = __builtin_amdgcn_wmma_f32_16x16x32_bf16(
            false, af[ms], false, bf[ns], (short)0, acc[ms][ns], false, false);
    __syncthreads();                              // done reading buf before overwrite
  }

  const int rowoff = kh << 3;
#pragma unroll
  for (int ms = 0; ms < 4; ++ms)
#pragma unroll
    for (int ns = 0; ns < 2; ++ns) {
      int colg = n0 + wn * 32 + ns * 16 + lr;
#pragma unroll
      for (int r = 0; r < 8; ++r) {
        int rowg = m0 + wm * 64 + ms * 16 + rowoff + r;
        float v = acc[ms][ns][r];
        if (BIAS)  v += bias[colg];
        if (GELU_) v = 0.5f * v * (1.0f + erff(v * 0.70710678118654752f));
        if (RES)   v += Res[(size_t)rowg * N + colg];
        if (OUTBF) Dbf[(size_t)rowg * N + colg] = (__bf16)v;
        else       D[(size_t)rowg * N + colg] = v;
      }
    }
}

// ---------------- flash attention (causal, WMMA, TDM K/V staging) -------------
// qkv: (ROWS, 3072) bf16 as [3][NH][HD]; o: (ROWS, 1024) bf16.
__global__ __launch_bounds__(256)
void attn_kernel(const __bf16* __restrict__ qkv, __bf16* __restrict__ o) {
  __shared__ alignas(16) __bf16 Kt[32][72];        // TDM pads 128B rows to 144B
  __shared__ alignas(16) __bf16 Vt[32][72];
  __shared__ alignas(16) __bf16 Ps[8][16][40];     // per-wave P staging [q][key]
  const int tid = threadIdx.x, lane = tid & 31, w = tid >> 5;
  const int lr = lane & 15, kh = lane >> 4;
  const int b = blockIdx.x >> 4, hh = blockIdx.x & 15;
  const int q0 = blockIdx.y * 128;
  const int qw = q0 + w * 16;
  const float NEG = -3.0e38f;

  union BF { v16bf v; __bf16 e[16]; };
  BF qf[2];
  {
    const __bf16* qrow = qkv + ((size_t)(b * T_ + qw + lr)) * 3072 + hh * 64;
#pragma unroll
    for (int f = 0; f < 2; ++f)
#pragma unroll
      for (int j = 0; j < 16; ++j) {
        int kd = f * 32 + ((j < 8) ? (kh * 8 + j) : (16 + kh * 8 + (j - 8)));
        qf[f].e[j] = (__bf16)((float)qrow[kd] * 0.125f);  // HD^-0.5 folded into Q
      }
  }

  v8f zero = {0.f, 0.f, 0.f, 0.f, 0.f, 0.f, 0.f, 0.f};
  float m_i[8], l_i[8];
  v8f accO[4];
#pragma unroll
  for (int r = 0; r < 8; ++r) { m_i[r] = NEG; l_i[r] = 0.f; }
#pragma unroll
  for (int n = 0; n < 4; ++n) accO[n] = zero;

  const uint ldsK = (uint)(uintptr_t)&Kt[0][0];
  const uint ldsV = (uint)(uintptr_t)&Vt[0][0];
  const int nkb = (q0 + 128) >> 5;                 // causal K/V span for this chunk
  for (int kb = 0; kb < nkb; ++kb) {
    __syncthreads();                               // all readers of Kt/Vt done
    if (w == 0) {                                  // TDM: stage 32 keys x 64 dims
      const __bf16* kg = qkv + ((size_t)(b * T_ + kb * 32)) * 3072 + 1024 + hh * 64;
      tdm_load_2d(ldsK, kg,        64, 32, 64, 32, 3072, 4, 3);
      tdm_load_2d(ldsV, kg + 1024, 64, 32, 64, 32, 3072, 4, 3);
      __builtin_amdgcn_s_wait_tensorcnt(0);
    }
    __syncthreads();
    if (kb * 32 <= qw + 15) {                      // wave-uniform predicate
      v8f s[2];
#pragma unroll
      for (int nh2 = 0; nh2 < 2; ++nh2) {          // S = Q @ K^T (two K-dim halves)
        v16bf k0f = load_frag_b(&Kt[nh2 * 16 + lr][0], kh);
        v16bf k1f = load_frag_b(&Kt[nh2 * 16 + lr][32], kh);
        v8f sv = zero;
        sv = __builtin_amdgcn_wmma_f32_16x16x32_bf16(false, qf[0].v, false, k0f, (short)0, sv, false, false);
        sv = __builtin_amdgcn_wmma_f32_16x16x32_bf16(false, qf[1].v, false, k1f, (short)0, sv, false, false);
        s[nh2] = sv;
      }
      const int rowoff = kh << 3;
#pragma unroll
      for (int nh2 = 0; nh2 < 2; ++nh2) {          // causal mask
        int tk = kb * 32 + nh2 * 16 + lr;
#pragma unroll
        for (int r = 0; r < 8; ++r)
          if (tk > qw + rowoff + r) s[nh2][r] = NEG;
      }
      float mx[8];
#pragma unroll
      for (int r = 0; r < 8; ++r) mx[r] = fmaxf(s[0][r], s[1][r]);
#pragma unroll
      for (int off = 1; off < 16; off <<= 1)
#pragma unroll
        for (int r = 0; r < 8; ++r) mx[r] = fmaxf(mx[r], __shfl_xor(mx[r], off, 32));

      float p0[8], p1[8];
#pragma unroll
      for (int r = 0; r < 8; ++r) {                // online softmax rescale
        float nm = fmaxf(m_i[r], mx[r]);
        float al = __expf(m_i[r] - nm);
        m_i[r] = nm;
        l_i[r] *= al;
#pragma unroll
        for (int n = 0; n < 4; ++n) accO[n][r] *= al;
        p0[r] = __expf(s[0][r] - nm);
        p1[r] = __expf(s[1][r] - nm);
      }
      float rs[8];
#pragma unroll
      for (int r = 0; r < 8; ++r) rs[r] = p0[r] + p1[r];
#pragma unroll
      for (int off = 1; off < 16; off <<= 1)
#pragma unroll
        for (int r = 0; r < 8; ++r) rs[r] += __shfl_xor(rs[r], off, 32);
#pragma unroll
      for (int r = 0; r < 8; ++r) l_i[r] += rs[r];

#pragma unroll
      for (int r = 0; r < 8; ++r) {                // C-layout -> A-layout via LDS
        int m = rowoff + r;
        Ps[w][m][lr]      = (__bf16)p0[r];
        Ps[w][m][16 + lr] = (__bf16)p1[r];
      }
      asm volatile("s_wait_dscnt 0" ::: "memory"); // wave-local LDS RAW fence

      v16bf pf = load_frag_a(&Ps[w][lr][0], kh);
#pragma unroll
      for (int ns = 0; ns < 4; ++ns) {             // O += P @ V
        BF vf;
#pragma unroll
        for (int j = 0; j < 16; ++j) vf.e[j] = Vt[(j < 8) ? (kh * 8 + j) : (16 + kh * 8 + (j - 8))][ns * 16 + lr];
        accO[ns] = __builtin_amdgcn_wmma_f32_16x16x32_bf16(
            false, pf, false, vf.v, (short)0, accO[ns], false, false);
      }
    }
  }

  const int rowoff = kh << 3;
#pragma unroll
  for (int ns = 0; ns < 4; ++ns)
#pragma unroll
    for (int r = 0; r < 8; ++r) {
      int tq = qw + rowoff + r;
      float val = accO[ns][r] / l_i[r];
      o[((size_t)(b * T_ + tq)) * 1024 + hh * 64 + ns * 16 + lr] = (__bf16)val;
    }
}

// ---------------- launch -------------------------------------------------------
static inline int cdiv(long long a, int b) { return (int)((a + b - 1) / b); }

extern "C" void kernel_launch(void* const* d_in, const int* in_sizes, int n_in,
                              void* d_out, int out_size, void* d_ws, size_t ws_size,
                              hipStream_t stream) {
  const float* z    = (const float*)d_in[0];
  const float* a    = (const float*)d_in[1];
  const float* te   = (const float*)d_in[2];
  const float* ipw  = (const float*)d_in[3];
  const float* ipb  = (const float*)d_in[4];
  const float* pos  = (const float*)d_in[5];
  const float* ln1w = (const float*)d_in[6];
  const float* ln1b = (const float*)d_in[7];
  const float* qkvw = (const float*)d_in[8];
  const float* projw= (const float*)d_in[9];
  const float* projb= (const float*)d_in[10];
  const float* ln2w = (const float*)d_in[11];
  const float* ln2b = (const float*)d_in[12];
  const float* w1   = (const float*)d_in[13];
  const float* b1   = (const float*)d_in[14];
  const float* w2   = (const float*)d_in[15];
  const float* b2   = (const float*)d_in[16];
  const float* onw  = (const float*)d_in[17];
  const float* onb  = (const float*)d_in[18];
  const float* dw   = (const float*)d_in[19];
  const float* db   = (const float*)d_in[20];

  // workspace: fp32 x, then bf16 activations + bf16 transposed weights
  char* p = (char*)d_ws;
  float*  x     = (float*)p;  p += (size_t)ROWS * 1024 * 4;
  __bf16* xin   = (__bf16*)p; p += (size_t)ROWS * 672 * 2;
  __bf16* h     = (__bf16*)p; p += (size_t)ROWS * 1024 * 2;
  __bf16* qkv   = (__bf16*)p; p += (size_t)ROWS * 3072 * 2;
  __bf16* attno = (__bf16*)p; p += (size_t)ROWS * 1024 * 2;
  __bf16* mid   = (__bf16*)p; p += (size_t)ROWS * 4096 * 2;
  __bf16* ipT   = (__bf16*)p; p += (size_t)672 * 1024 * 2;
  __bf16* qkvT  = (__bf16*)p; p += (size_t)L_ * 1024 * 3072 * 2;
  __bf16* projT = (__bf16*)p; p += (size_t)L_ * 1024 * 1024 * 2;
  __bf16* w1T   = (__bf16*)p; p += (size_t)L_ * 1024 * 4096 * 2;
  __bf16* w2T   = (__bf16*)p; p += (size_t)L_ * 4096 * 1024 * 2;
  __bf16* dT    = (__bf16*)p;

  // weight convert+transpose (bf16, [N][K])
  wtrans_kernel<<<cdiv((long long)672 * 1024, 256), 256, 0, stream>>>(ipw, ipT, 672, 1024);
  for (int l = 0; l < L_; ++l) {
    wtrans_kernel<<<cdiv((long long)1024 * 3072, 256), 256, 0, stream>>>(
        qkvw + (size_t)l * 1024 * 3072, qkvT + (size_t)l * 3072 * 1024, 1024, 3072);
    wtrans_kernel<<<cdiv((long long)1024 * 1024, 256), 256, 0, stream>>>(
        projw + (size_t)l * 1024 * 1024, projT + (size_t)l * 1024 * 1024, 1024, 1024);
    wtrans_kernel<<<cdiv((long long)1024 * 4096, 256), 256, 0, stream>>>(
        w1 + (size_t)l * 1024 * 4096, w1T + (size_t)l * 4096 * 1024, 1024, 4096);
    wtrans_kernel<<<cdiv((long long)4096 * 1024, 256), 256, 0, stream>>>(
        w2 + (size_t)l * 4096 * 1024, w2T + (size_t)l * 1024 * 4096, 4096, 1024);
  }
  wtrans_kernel<<<cdiv((long long)1024 * 512, 256), 256, 0, stream>>>(dw, dT, 1024, 512);

  concat_kernel<<<21504, 256, 0, stream>>>(z, a, te, xin);
  gemm_kernel<1, 0, 0, 0><<<dim3(8, 64), 256, 0, stream>>>(
      xin, ipT, ipb, nullptr, x, nullptr, ROWS, 1024, 672);
  addpos_kernel<<<32768, 256, 0, stream>>>(x, pos);

  for (int l = 0; l < L_; ++l) {
    ln_kernel<<<ROWS, 256, 0, stream>>>(x, ln1w + l * 1024, ln1b + l * 1024, h);
    gemm_kernel<0, 0, 0, 1><<<dim3(24, 64), 256, 0, stream>>>(
        h, qkvT + (size_t)l * 3072 * 1024, nullptr, nullptr, nullptr, qkv, ROWS, 3072, 1024);
    attn_kernel<<<dim3(256, 4), 256, 0, stream>>>(qkv, attno);
    gemm_kernel<1, 1, 0, 0><<<dim3(8, 64), 256, 0, stream>>>(
        attno, projT + (size_t)l * 1024 * 1024, projb + l * 1024, x, x, nullptr, ROWS, 1024, 1024);
    ln_kernel<<<ROWS, 256, 0, stream>>>(x, ln2w + l * 1024, ln2b + l * 1024, h);
    gemm_kernel<1, 0, 1, 1><<<dim3(32, 64), 256, 0, stream>>>(
        h, w1T + (size_t)l * 4096 * 1024, b1 + l * 4096, nullptr, nullptr, mid, ROWS, 4096, 1024);
    gemm_kernel<1, 1, 0, 0><<<dim3(8, 64), 256, 0, stream>>>(
        mid, w2T + (size_t)l * 1024 * 4096, b2 + l * 1024, x, x, nullptr, ROWS, 1024, 4096);
  }

  ln_kernel<<<ROWS, 256, 0, stream>>>(x, onw, onb, h);
  gemm_kernel<1, 1, 0, 0><<<dim3(4, 64), 256, 0, stream>>>(
      h, dT, db, z, (float*)d_out, nullptr, ROWS, 512, 1024);
}
